// DensePromptGenerator_29557964931400
// MI455X (gfx1250) — matrix operand order repaired and checked by reading, not compiled, not run on hardware
//
#include <hip/hip_runtime.h>
#include <hip/hip_bf16.h>

#define B_  16
#define C_  256
#define H_  64
#define W_  64
#define HW_ 4096
#define Q_  8
#define N_  9

typedef __attribute__((ext_vector_type(16))) __bf16          v16bf;
typedef __attribute__((ext_vector_type(16))) unsigned short  v16us;
typedef __attribute__((ext_vector_type(8)))  float           v8f;

__device__ __forceinline__ unsigned short f2bf(float f) {
    unsigned u = __float_as_uint(f);
    unsigned r = u + 0x7FFFu + ((u >> 16) & 1u);   // round-to-nearest-even
    return (unsigned short)(r >> 16);
}

__device__ __forceinline__ float gelu_f(float x) {
    return 0.5f * x * (1.0f + erff(x * 0.70710678118654752f));
}

__device__ __forceinline__ float block_reduce_sum(float v, float* red, int tid) {
    red[tid] = v;
    __syncthreads();
    for (int s = 128; s > 0; s >>= 1) {
        if (tid < s) red[tid] += red[tid + s];
        __syncthreads();
    }
    float r = red[0];
    __syncthreads();
    return r;
}

// ---------------------------------------------------------------------------
// 1. Prototype token path: pt = inter @ intra^T, t = LN(gelu(cat @ proj) @ lin)
// ---------------------------------------------------------------------------
__global__ __launch_bounds__(256)
void token_kernel(const float* __restrict__ inter, const float* __restrict__ intra,
                  const float* __restrict__ proj_w, const float* __restrict__ proj_b,
                  const float* __restrict__ lin_w, const float* __restrict__ lin_b,
                  const float* __restrict__ tok_g, const float* __restrict__ tok_b,
                  float* __restrict__ t_out) {
    const int bq = blockIdx.x;
    const int b = bq / Q_, q = bq % Q_;
    const int tid = threadIdx.x;
    __shared__ float vin[C_ + N_];
    __shared__ float mid[C_];
    __shared__ float red[256];

    vin[tid] = inter[(size_t)(b * Q_ + q) * C_ + tid];
    __syncthreads();
    if (tid < N_) {
        float s = 0.f;
        const float* ir = intra + (size_t)(b * N_ + tid) * C_;
        for (int c = 0; c < C_; ++c) s += ir[c] * vin[c];
        vin[C_ + tid] = s;
    }
    __syncthreads();

    float t1 = proj_b[tid];
    for (int j = 0; j < C_ + N_; ++j) t1 += vin[j] * proj_w[(size_t)j * C_ + tid];
    mid[tid] = gelu_f(t1);
    __syncthreads();

    float t2 = lin_b[tid];
    for (int j = 0; j < C_; ++j) t2 += mid[j] * lin_w[(size_t)j * C_ + tid];

    float s1 = block_reduce_sum(t2, red, tid);
    float s2 = block_reduce_sum(t2 * t2, red, tid);
    float m = s1 * (1.0f / C_);
    float var = s2 * (1.0f / C_) - m * m;
    t_out[(size_t)(b * Q_ + q) * C_ + tid] =
        (t2 - m) * rsqrtf(var + 1e-5f) * tok_g[tid] + tok_b[tid];
}

// ---------------------------------------------------------------------------
// 2. Mask gating: img = image * (sum_n intra[b,n,c]*masks[b,n,hw]) + image
// ---------------------------------------------------------------------------
__global__ __launch_bounds__(256)
void gate_kernel(const float* __restrict__ image, const float* __restrict__ intra,
                 const float* __restrict__ masks, float* __restrict__ img) {
    const int hw = blockIdx.x, b = blockIdx.y, c = threadIdx.x;
    float pe = 0.f;
#pragma unroll
    for (int n = 0; n < N_; ++n)
        pe += intra[(size_t)(b * N_ + n) * C_ + c] * masks[(size_t)(b * N_ + n) * HW_ + hw];
    size_t idx = ((size_t)b * C_ + c) * HW_ + hw;
    float v = image[idx];
    img[idx] = v * pe + v;
}

// ---------------------------------------------------------------------------
// 3. 1x1 conv as WMMA GEMM: Y[b,co,hw] = sum_ci W[co,ci] * X[b,ci,hw] + bias
//    Per batch: A = W (MxK, row-major), B = X (KxN, rows contiguous in hw).
//    Each wave: one 16(co) x 32(hw) strip = 2 WMMA tiles sharing the A frag.
//    Block of 8 waves covers 32(co) x 128(hw). CIN templated -> full unroll.
// ---------------------------------------------------------------------------
template <int CIN, bool GELU, bool RES>
__global__ __launch_bounds__(256)
void conv1x1_wmma_kernel(const float* __restrict__ X, const float* __restrict__ Wt,
                         const float* __restrict__ bias, const float* __restrict__ residual,
                         float* __restrict__ Y, int Cout) {
    const int b = blockIdx.z;
    const int tid = threadIdx.x;
    const int wave = tid >> 5, lane = tid & 31;
    const int half = lane >> 4, lr = lane & 15;
    const int mt = wave >> 2, nt = wave & 3;
    const int co_base = blockIdx.y * 32 + mt * 16;
    const int hw_base = blockIdx.x * 128 + nt * 32;
    const int co  = co_base + lr;    // A-fragment row this lane feeds
    const int hw0 = hw_base + lr;    // B-fragment column, tile 0
    const int hw1 = hw0 + 16;        // B-fragment column, tile 1
    const float* Xb = X + (size_t)b * CIN * HW_;
    const float* Wr = Wt + (size_t)co * CIN;

    v8f acc0 = {}, acc1 = {};
#pragma unroll
    for (int kk = 0; kk < CIN; kk += 32) {
        v16us au, bu0, bu1;
#pragma unroll
        for (int i = 0; i < 8; ++i) {
            // A (16x32, 16-bit): VGPR i holds K = (i/4)*16 + half*8 + (i%4)*2 (+1)
            int ka = kk + ((i >> 2) << 4) + (half << 3) + ((i & 3) << 1);
            float2 wv = *(const float2*)(Wr + ka);
            au[2 * i]     = f2bf(wv.x);
            au[2 * i + 1] = f2bf(wv.y);
            // B (32x16, 16-bit): VGPR i holds K = (i/4)*16 + half*8 + (i%4)*2 (+1), same as A
            int kb = kk + ((i >> 2) << 4) + (half << 3) + ((i & 3) << 1);
            const float* r0 = Xb + (size_t)kb * HW_;
            const float* r1 = Xb + (size_t)(kb + 1) * HW_;
            bu0[2 * i]     = f2bf(r0[hw0]);
            bu0[2 * i + 1] = f2bf(r1[hw0]);
            bu1[2 * i]     = f2bf(r0[hw1]);
            bu1[2 * i + 1] = f2bf(r1[hw1]);
            if (kk + 32 < CIN) {  // pull next K-block toward caches (global_prefetch_b8)
                __builtin_prefetch(r0 + 32 * HW_ + hw0, 0, 0);
            }
        }
        v16bf a  = __builtin_bit_cast(v16bf, au);
        v16bf b0 = __builtin_bit_cast(v16bf, bu0);
        v16bf b1 = __builtin_bit_cast(v16bf, bu1);
        acc0 = __builtin_amdgcn_wmma_f32_16x16x32_bf16(false, a, false, b0, (short)0, acc0, false, false);
        acc1 = __builtin_amdgcn_wmma_f32_16x16x32_bf16(false, a, false, b1, (short)0, acc1, false, false);
    }

    float* Yb = Y + (size_t)b * Cout * HW_;
#pragma unroll
    for (int r = 0; r < 8; ++r) {
        int com = co_base + half * 8 + r;      // C/D layout: m = r + (lane/16)*8
        size_t row = (size_t)com * HW_;
        float v0 = acc0[r] + bias[com];
        float v1 = acc1[r] + bias[com];
        if (RES) {
            const float* Rb = residual + (size_t)b * Cout * HW_ + row;
            v0 += Rb[hw0];
            v1 += Rb[hw1];
        }
        if (GELU) { v0 = gelu_f(v0); v1 = gelu_f(v1); }
        Yb[row + hw0] = v0;
        Yb[row + hw1] = v1;
    }
}

// ---------------------------------------------------------------------------
// 4. Offset conv: 3x3, C->27, pad=dil, dilation=dil (direct)
// ---------------------------------------------------------------------------
__global__ __launch_bounds__(432)
void offconv_kernel(const float* __restrict__ X, const float* __restrict__ ow,
                    const float* __restrict__ ob, float* __restrict__ off, int dil) {
    const int hw = blockIdx.x * 16 + threadIdx.x;
    const int b = blockIdx.y, oc = threadIdx.y;
    const int h = hw >> 6, w = hw & 63;
    float s = ob[oc];
    const float* Xb = X + (size_t)b * C_ * HW_;
#pragma unroll
    for (int ki = 0; ki < 3; ++ki) {
#pragma unroll
        for (int kj = 0; kj < 3; ++kj) {
            int yy = h + (ki - 1) * dil, xx = w + (kj - 1) * dil;
            if (yy < 0 || yy >= H_ || xx < 0 || xx >= W_) continue;
            const float* xp = Xb + yy * W_ + xx;
            const float* wp = ow + ((size_t)oc * C_ * 9) + (ki * 3 + kj);
            for (int ci = 0; ci < C_; ++ci)
                s += xp[(size_t)ci * HW_] * wp[(size_t)ci * 9];
        }
    }
    off[((size_t)b * 27 + oc) * HW_ + hw] = s;
}

// ---------------------------------------------------------------------------
// 5. Modulated deformable depthwise 3x3 apply
// ---------------------------------------------------------------------------
__global__ __launch_bounds__(256)
void deform_kernel(const float* __restrict__ X, const float* __restrict__ off,
                   const float* __restrict__ dww, const float* __restrict__ dwb,
                   float* __restrict__ out, int dil) {
    const int hw = blockIdx.x, b = blockIdx.y, c = threadIdx.x;
    const int h = hw >> 6, w = hw & 63;
    __shared__ float sOff[27];
    if (c < 27) sOff[c] = off[((size_t)b * 27 + c) * HW_ + hw];
    __syncthreads();

    const float* Xc = X + ((size_t)b * C_ + c) * HW_;
    float acc = 0.f;
#pragma unroll
    for (int kk = 0; kk < 9; ++kk) {
        int ki = kk / 3, kj = kk % 3;
        float oxv = sOff[kk];
        float oyv = sOff[9 + kk];
        float mv = 1.0f / (1.0f + expf(-sOff[18 + kk]));
        float py = (float)(h - dil + ki * dil) + oyv;
        float px = (float)(w - dil + kj * dil) + oxv;
        float y0f = floorf(py), x0f = floorf(px);
        float wy = py - y0f, wx = px - x0f;
        int y0 = (int)y0f, x0 = (int)x0f;
        float s = 0.f;
#pragma unroll
        for (int dy = 0; dy < 2; ++dy) {
#pragma unroll
            for (int dx = 0; dx < 2; ++dx) {
                int yi = y0 + dy, xi = x0 + dx;
                float wgt = (dy ? wy : 1.f - wy) * (dx ? wx : 1.f - wx);
                if (yi >= 0 && yi < H_ && xi >= 0 && xi < W_)
                    s += Xc[yi * W_ + xi] * wgt;
            }
        }
        acc += s * mv * dww[c * 9 + kk];
    }
    out[((size_t)b * C_ + c) * HW_ + hw] = acc + dwb[c];
}

// ---------------------------------------------------------------------------
// 6. Fused x = x*a + img, then LayerNorm over channel dim
// ---------------------------------------------------------------------------
__global__ __launch_bounds__(256)
void fuse_ln_kernel(const float* __restrict__ X, const float* __restrict__ A,
                    const float* __restrict__ IMG, const float* __restrict__ g,
                    const float* __restrict__ bb, float* __restrict__ out) {
    const int hw = blockIdx.x, b = blockIdx.y, c = threadIdx.x;
    __shared__ float red[256];
    size_t idx = ((size_t)b * C_ + c) * HW_ + hw;
    float v = X[idx] * A[idx] + IMG[idx];
    float s1 = block_reduce_sum(v, red, c);
    float s2 = block_reduce_sum(v * v, red, c);
    float m = s1 * (1.0f / C_);
    float var = s2 * (1.0f / C_) - m * m;
    out[idx] = (v - m) * rsqrtf(var + 1e-5f) * g[c] + bb[c];
}

// ---------------------------------------------------------------------------
// 7. Depthwise 3x3 (pad=1) + GELU
// ---------------------------------------------------------------------------
__global__ __launch_bounds__(256)
void dw3_gelu_kernel(const float* __restrict__ X, const float* __restrict__ w3,
                     const float* __restrict__ b3, float* __restrict__ out) {
    const int hw = blockIdx.x, b = blockIdx.y, c = threadIdx.x;
    const int h = hw >> 6, w = hw & 63;
    const float* Xc = X + ((size_t)b * C_ + c) * HW_;
    float s = b3[c];
#pragma unroll
    for (int ki = 0; ki < 3; ++ki) {
#pragma unroll
        for (int kj = 0; kj < 3; ++kj) {
            int yy = h + ki - 1, xx = w + kj - 1;
            if (yy < 0 || yy >= H_ || xx < 0 || xx >= W_) continue;
            s += Xc[yy * W_ + xx] * w3[c * 9 + ki * 3 + kj];
        }
    }
    out[((size_t)b * C_ + c) * HW_ + hw] = gelu_f(s);
}

// ---------------------------------------------------------------------------
// 8. 8-token cross-attention: softmax_q(dense . t * scale), recombine + alpha
// ---------------------------------------------------------------------------
__global__ __launch_bounds__(256)
void attn_kernel(const float* __restrict__ dense, const float* __restrict__ t,
                 const float* __restrict__ alpha, float* __restrict__ out) {
    const int hw = blockIdx.x, b = blockIdx.y, c = threadIdx.x;
    __shared__ float tS[Q_ * C_];
    __shared__ float red[256];
    for (int i = c; i < Q_ * C_; i += 256) tS[i] = t[(size_t)b * Q_ * C_ + i];
    __syncthreads();

    size_t idx = ((size_t)b * C_ + c) * HW_ + hw;
    float dv = dense[idx];
    float lg[Q_];
#pragma unroll
    for (int q = 0; q < Q_; ++q)
        lg[q] = block_reduce_sum(dv * tS[q * C_ + c], red, c) * 0.0625f;  // C^-0.5
    float mx = lg[0];
#pragma unroll
    for (int q = 1; q < Q_; ++q) mx = fmaxf(mx, lg[q]);
    float den = 0.f;
#pragma unroll
    for (int q = 0; q < Q_; ++q) { lg[q] = expf(lg[q] - mx); den += lg[q]; }
    float o = 0.f;
#pragma unroll
    for (int q = 0; q < Q_; ++q) o += (lg[q] / den) * tS[q * C_ + c];
    out[idx] = o + dv * alpha[c];
}

// ---------------------------------------------------------------------------
extern "C" void kernel_launch(void* const* d_in, const int* in_sizes, int n_in,
                              void* d_out, int out_size, void* d_ws, size_t ws_size,
                              hipStream_t stream) {
    const float* image  = (const float*)d_in[0];
    const float* inter  = (const float*)d_in[1];
    const float* intra  = (const float*)d_in[2];
    const float* masks  = (const float*)d_in[3];
    const float* proj_w = (const float*)d_in[4];
    const float* proj_b = (const float*)d_in[5];
    const float* lin_w  = (const float*)d_in[6];
    const float* lin_b  = (const float*)d_in[7];
    const float* tok_g  = (const float*)d_in[8];
    const float* tok_b  = (const float*)d_in[9];
    const float* alpha  = (const float*)d_in[10];
    const float* in_w   = (const float*)d_in[11];
    const float* in_b   = (const float*)d_in[12];
    const float* off0_w = (const float*)d_in[13];
    const float* off0_b = (const float*)d_in[14];
    const float* dw0_w  = (const float*)d_in[15];
    const float* dw0_b  = (const float*)d_in[16];
    const float* off1_w = (const float*)d_in[17];
    const float* off1_b = (const float*)d_in[18];
    const float* dw1_w  = (const float*)d_in[19];
    const float* dw1_b  = (const float*)d_in[20];
    const float* cv_w   = (const float*)d_in[21];
    const float* cv_b   = (const float*)d_in[22];
    const float* ln_g   = (const float*)d_in[23];
    const float* ln_b   = (const float*)d_in[24];
    const float* op0_w  = (const float*)d_in[25];
    const float* op0_b  = (const float*)d_in[26];
    const float* op1_w  = (const float*)d_in[27];
    const float* op1_b  = (const float*)d_in[28];
    const float* op2_w  = (const float*)d_in[29];
    const float* op2_b  = (const float*)d_in[30];
    const float* out0_w = (const float*)d_in[31];
    const float* out0_b = (const float*)d_in[32];
    const float* out1_w = (const float*)d_in[33];
    const float* out1_b = (const float*)d_in[34];

    const size_t SZ = (size_t)B_ * C_ * HW_;   // 16.7M floats per full tensor
    float* ws   = (float*)d_ws;
    float* buf0 = ws;                                      // img / y / attn-out
    float* buf1 = ws + SZ;                                 // x / dw-out / half-out
    float* buf2 = ws + 2 * SZ;                             // a0 / a / dense
    float* offb = ws + 3 * SZ;                             // [B,27,HW]
    float* tbuf = ws + 3 * SZ + (size_t)B_ * 27 * HW_;     // [B,Q,C]
    float* xn   = (float*)d_out;                           // a1 / xn; dead before out1

    dim3 bpix(256), gpix(HW_, B_);
    dim3 goff(HW_ / 16, B_), boff(16, 27);
    dim3 g256(HW_ / 128, C_ / 32, B_);       // Cout=256 GEMM grid
    dim3 g128(HW_ / 128, (C_ / 2) / 32, B_); // Cout=128 GEMM grid

    // tokens
    token_kernel<<<dim3(B_ * Q_), 256, 0, stream>>>(
        inter, intra, proj_w, proj_b, lin_w, lin_b, tok_g, tok_b, tbuf);
    // img = image * pe + image
    gate_kernel<<<gpix, bpix, 0, stream>>>(image, intra, masks, buf0);
    // x = gelu(conv1x1(img))
    conv1x1_wmma_kernel<C_, true, false><<<g256, 256, 0, stream>>>(
        buf0, in_w, in_b, nullptr, buf1, C_);
    // deform block 0 (dil=1): x -> a0
    offconv_kernel<<<goff, boff, 0, stream>>>(buf1, off0_w, off0_b, offb, 1);
    deform_kernel<<<gpix, bpix, 0, stream>>>(buf1, offb, dw0_w, dw0_b, buf2, 1);
    // deform block 1 (dil=3): a0 -> a1 (in d_out, dead later)
    offconv_kernel<<<goff, boff, 0, stream>>>(buf2, off1_w, off1_b, offb, 3);
    deform_kernel<<<gpix, bpix, 0, stream>>>(buf2, offb, dw1_w, dw1_b, xn, 3);
    // a = conv1x1(a1)
    conv1x1_wmma_kernel<C_, false, false><<<g256, 256, 0, stream>>>(
        xn, cv_w, cv_b, nullptr, buf2, C_);
    // xn = LN(x*a + img)
    fuse_ln_kernel<<<gpix, bpix, 0, stream>>>(buf1, buf2, buf0, ln_g, ln_b, xn);
    // dense = conv1x1(gelu(dw3(conv1x1(xn)))) + xn
    conv1x1_wmma_kernel<C_, false, false><<<g256, 256, 0, stream>>>(
        xn, op0_w, op0_b, nullptr, buf0, C_);
    dw3_gelu_kernel<<<gpix, bpix, 0, stream>>>(buf0, op1_w, op1_b, buf1);
    conv1x1_wmma_kernel<C_, false, true><<<g256, 256, 0, stream>>>(
        buf1, op2_w, op2_b, xn, buf2, C_);
    // cross-attention over 8 tokens
    attn_kernel<<<gpix, bpix, 0, stream>>>(buf2, tbuf, alpha, buf0);
    // output head
    conv1x1_wmma_kernel<C_, true, false><<<g128, 256, 0, stream>>>(
        buf0, out0_w, out0_b, nullptr, buf1, C_ / 2);
    conv1x1_wmma_kernel<C_ / 2, false, false><<<g256, 256, 0, stream>>>(
        buf1, out1_w, out1_b, nullptr, (float*)d_out, C_);
}
